// RNNFC_395136991535
// MI455X (gfx1250) — matrix operand order repaired and checked by reading, not compiled, not run on hardware
//
#include <hip/hip_runtime.h>

// RNN + FC readout for MI455X (gfx1250), wave32, WMMA bf16 path.
//
//   phase 0: convert w_ih / w_hh / w_out f32 -> bf16 (one tiny kernel)
//   phase 1: x_proj[BT,512] = input[BT,256] @ w_ih^T + b_h      (WMMA bf16, f32 acc)
//   phase 2: h_t = tanh(xp_t + h_{t-1} @ w_hh^T), single WGP.
//            h resident in LDS, w_hh streamed from L2, x_proj staged into LDS with
//            global_load_async_to_lds_b128, h_t drained with
//            global_store_async_from_lds_b128 (ASYNCcnt pipelined across steps).
//   phase 3: out[BT,256] = hs[BT,512] @ w_out^T + b_out         (WMMA bf16)

typedef __attribute__((ext_vector_type(16))) __bf16 v16bf;
typedef __attribute__((ext_vector_type(8)))  float  v8f;
typedef __attribute__((ext_vector_type(4)))  float  vf4;

#define IN_SIZE  256
#define HID      512
#define OUT_SIZE 256
#define BATCH    64
#define NSTEPS   2048
#define BT       (BATCH * NSTEPS)   // 131072 rows
#define LDH      520                // h row stride (bf16): 1040B -> 4-bank shift per row
#define XPLD     516                // xp row stride (f32): lane halves 32 banks apart
#define HSH_BYTES (BATCH * LDH * 2) // 66560
#define XPS_BYTES (BATCH * XPLD * 4)// 132096

union Frag {
    v16bf  v;
    __bf16 h[16];
    vf4    q[2];   // two 16-byte chunks
};

// ---------------- WMMA fragment helpers (layouts per CDNA5 ISA §7.12.2) ----------------

// 16-bit A-matrix 16x32: lane L -> row m0+(L&15); two contiguous 8-half runs at
// K = k0+kb.. and K = k0+16+kb.., kb = (L>>4)*8.
__device__ inline v16bf load_A_bf16(const __bf16* base, int ldk, int m0, int k0) {
    const int lane = threadIdx.x & 31;
    const int m    = m0 + (lane & 15);
    const int kb   = (lane >> 4) * 8;
    const __bf16* p = base + (size_t)m * ldk + k0 + kb;
    Frag f;
    f.q[0] = *reinterpret_cast<const vf4*>(p);
    f.q[1] = *reinterpret_cast<const vf4*>(p + 16);
    return f.v;
}

// Same A fragment sourced from f32 memory with on-the-fly bf16 conversion.
__device__ inline v16bf load_A_f32(const float* base, int ldk, int m0, int k0) {
    const int lane = threadIdx.x & 31;
    const int m    = m0 + (lane & 15);
    const int kb   = (lane >> 4) * 8;
    const float* p = base + (size_t)m * ldk + k0 + kb;
    Frag f;
#pragma unroll
    for (int i = 0; i < 8; ++i) f.h[i]     = (__bf16)p[i];
#pragma unroll
    for (int i = 0; i < 8; ++i) f.h[8 + i] = (__bf16)p[16 + i];
    return f.v;
}

// 16-bit B-matrix 32x16: lane L -> column n0+(L&15); two contiguous 8-half runs at
// K = k0+kb.. and K = k0+16+kb.., kb = (L>>4)*8.
// For row-major W[N][K] (X @ W^T) this is W[n][ks..ks+7] and W[n][ks+16..ks+23].
__device__ inline v16bf load_B_bf16(const __bf16* W, int ldk, int n0, int k0) {
    const int lane = threadIdx.x & 31;
    const int n    = n0 + (lane & 15);
    const int ks   = k0 + (lane >> 4) * 8;
    const __bf16* p = W + (size_t)n * ldk + ks;
    Frag f;
    f.q[0] = *reinterpret_cast<const vf4*>(p);
    f.q[1] = *reinterpret_cast<const vf4*>(p + 16);
    return f.v;
}

__device__ inline v8f wmma_bf16(v16bf a, v16bf b, v8f c) {
    return __builtin_amdgcn_wmma_f32_16x16x32_bf16(false, a, false, b, (short)0, c, false, false);
}

// Low 32 bits of the shared-aperture address == LDS byte offset (ISA §10.2).
__device__ inline unsigned lds_off32(const void* p) {
    return (unsigned)(unsigned long long)p;
}

// ---------------- phase 0: weight conversion ----------------

__global__ void cvt_weights(const float* __restrict__ w_ih, const float* __restrict__ w_hh,
                            const float* __restrict__ w_out,
                            __bf16* __restrict__ w_ih_b, __bf16* __restrict__ w_hh_b,
                            __bf16* __restrict__ w_out_b) {
    const int i = blockIdx.x * blockDim.x + threadIdx.x;
    if (i < HID * IN_SIZE)  w_ih_b[i]  = (__bf16)w_ih[i];
    if (i < HID * HID)      w_hh_b[i]  = (__bf16)w_hh[i];
    if (i < OUT_SIZE * HID) w_out_b[i] = (__bf16)w_out[i];
}

// ---------------- phase 1: x_proj = input @ w_ih^T + b_h ----------------

__global__ __launch_bounds__(256) void xproj_kernel(const float* __restrict__ input,
                                                    const __bf16* __restrict__ w_ih_b,
                                                    const float* __restrict__ b_h,
                                                    float* __restrict__ x_proj) {
    const int lane  = threadIdx.x & 31;
    const int gw    = blockIdx.x * 8 + (threadIdx.x >> 5);
    const int n0    = (gw & 31) * 16;        // 32 N tiles
    const int mquad = gw >> 5;               // 2048 quads of 64 rows

    Frag B[8];
#pragma unroll
    for (int kc = 0; kc < 8; ++kc) B[kc].v = load_B_bf16(w_ih_b, IN_SIZE, n0, kc * 32);

    const float bias = b_h[n0 + (lane & 15)];
#pragma unroll
    for (int mt = 0; mt < 4; ++mt) {
        const int m0 = mquad * 64 + mt * 16;
        v8f acc = {bias, bias, bias, bias, bias, bias, bias, bias};
#pragma unroll
        for (int kc = 0; kc < 8; ++kc)
            acc = wmma_bf16(load_A_f32(input, IN_SIZE, m0, kc * 32), B[kc].v, acc);
#pragma unroll
        for (int r = 0; r < 8; ++r) {
            const int m = m0 + r + 8 * (lane >> 4);
            x_proj[(size_t)m * HID + n0 + (lane & 15)] = acc[r];
        }
    }
}

// ---------------- phase 2: sequential recurrence, single workgroup ----------------
// 32 waves; wave w owns a 2x2 block of 16x16 tiles: mp = w&1 (rows mp*32..+31),
// np = w>>1 (cols np*32..+31). Balanced duplication: w_hh from L2 2x, h from LDS 16x
// (~1 MB/step each, matching the ~2048-clk/step WMMA floor).

__global__ __launch_bounds__(1024) void rnn_kernel(const float* __restrict__ x_proj,
                                                   const __bf16* __restrict__ w_hh_b,
                                                   __bf16* __restrict__ hs) {
    extern __shared__ char smem[];
    __bf16* hsh = (__bf16*)smem;                 // [BATCH][LDH]   h_t, bf16
    float*  xps = (float*)(smem + HSH_BYTES);    // [BATCH][XPLD]  x_proj slice, f32

    const int tid  = threadIdx.x;
    const int lane = tid & 31;
    const int wave = tid >> 5;
    const int m0   = (wave & 1) * 32;            // two M tiles: m0, m0+16
    const int n0   = (wave >> 1) * 32;           // two N tiles: n0, n0+16

    for (int i = tid; i < BATCH * LDH; i += 1024) hsh[i] = (__bf16)0.0f;   // h_0 = 0
    __syncthreads();

    for (int t = 0; t < NSTEPS; ++t) {
        // ---- stage this step's x_proj slice (64x512 f32 = 128 KB) into LDS, async.
        // 8192 16B chunks / 1024 threads = 8 per thread; overlapped with the WMMA loop.
#pragma unroll
        for (int k = 0; k < 8; ++k) {
            const int c   = tid + k * 1024;
            const int b   = c >> 7;              // 128 x 16B chunks per 512-f32 row
            const int off = (c & 127) << 2;      // f32 index, multiple of 4
            const float* gp = x_proj + ((size_t)b * NSTEPS + t) * HID + off;
            const unsigned lo = lds_off32(xps + b * XPLD + off);
            asm volatile("global_load_async_to_lds_b128 %0, %1, off"
                         :: "v"(lo), "v"((unsigned long long)gp) : "memory");
        }

        // ---- h_{t-1} @ w_hh^T over K=512
        v8f acc[4];
#pragma unroll
        for (int j = 0; j < 4; ++j) acc[j] = (v8f){0, 0, 0, 0, 0, 0, 0, 0};

#pragma unroll 4
        for (int kc = 0; kc < 16; ++kc) {
            const v16bf A0 = load_A_bf16(hsh, LDH, m0,      kc * 32);
            const v16bf A1 = load_A_bf16(hsh, LDH, m0 + 16, kc * 32);
            const v16bf B0 = load_B_bf16(w_hh_b, HID, n0,      kc * 32);
            const v16bf B1 = load_B_bf16(w_hh_b, HID, n0 + 16, kc * 32);
            acc[0] = wmma_bf16(A0, B0, acc[0]);
            acc[1] = wmma_bf16(A0, B1, acc[1]);
            acc[2] = wmma_bf16(A1, B0, acc[2]);
            acc[3] = wmma_bf16(A1, B1, acc[3]);
        }

        // xp staged + last step's hs drain done reading LDS; then publish across waves.
        asm volatile("s_wait_asynccnt 0" ::: "memory");
        __syncthreads();

        // ---- h_t = tanh(acc + xp), write into LDS h buffer
#pragma unroll
        for (int jm = 0; jm < 2; ++jm) {
#pragma unroll
            for (int jn = 0; jn < 2; ++jn) {
                const int n = n0 + jn * 16 + (lane & 15);
#pragma unroll
                for (int r = 0; r < 8; ++r) {
                    const int b = m0 + jm * 16 + r + 8 * (lane >> 4);
                    float v = acc[jm * 2 + jn][r] + xps[b * XPLD + n];
                    v = tanhf(v);
                    hsh[b * LDH + n] = (__bf16)v;
                }
            }
        }
        __syncthreads();   // h_t complete in LDS

        // ---- drain h_t -> hs[b*T+t][:] via async LDS->global (4 x 16B chunks/thread).
        // Completion is enforced by the s_wait_asynccnt after the next step's WMMA loop
        // (before h is overwritten); S_ENDPGM's implicit wait-idle covers the last step.
#pragma unroll
        for (int k = 0; k < 4; ++k) {
            const int c   = tid + k * 1024;
            const int b   = c >> 6;              // 64 x 16B chunks per 512-bf16 row
            const int off = (c & 63) << 3;       // bf16 index, multiple of 8
            __bf16* gp = hs + ((size_t)b * NSTEPS + t) * HID + off;
            const unsigned lo = lds_off32(hsh + b * LDH + off);
            asm volatile("global_store_async_from_lds_b128 %0, %1, off"
                         :: "v"((unsigned long long)gp), "v"(lo) : "memory");
        }
    }
}

// ---------------- phase 3: out = hs @ w_out^T + b_out ----------------

__global__ __launch_bounds__(256) void out_kernel(const __bf16* __restrict__ hs,
                                                  const __bf16* __restrict__ w_out_b,
                                                  const float* __restrict__ b_out,
                                                  float* __restrict__ out) {
    const int lane  = threadIdx.x & 31;
    const int gw    = blockIdx.x * 8 + (threadIdx.x >> 5);
    const int n0    = (gw & 15) * 16;        // 16 N tiles
    const int mquad = gw >> 4;               // 2048 quads of 64 rows

    Frag B[16];
#pragma unroll
    for (int kc = 0; kc < 16; ++kc) B[kc].v = load_B_bf16(w_out_b, HID, n0, kc * 32);

    const float bias = b_out[n0 + (lane & 15)];
#pragma unroll
    for (int mt = 0; mt < 4; ++mt) {
        const int m0 = mquad * 64 + mt * 16;
        v8f acc = {bias, bias, bias, bias, bias, bias, bias, bias};
#pragma unroll
        for (int kc = 0; kc < 16; ++kc)
            acc = wmma_bf16(load_A_bf16(hs, HID, m0, kc * 32), B[kc].v, acc);
#pragma unroll
        for (int r = 0; r < 8; ++r) {
            const int m = m0 + r + 8 * (lane >> 4);
            out[(size_t)m * OUT_SIZE + n0 + (lane & 15)] = acc[r];
        }
    }
}

// ---------------- host ----------------

extern "C" void kernel_launch(void* const* d_in, const int* in_sizes, int n_in,
                              void* d_out, int out_size, void* d_ws, size_t ws_size,
                              hipStream_t stream) {
    const float* input = (const float*)d_in[0];   // [64, 2048, 256]
    const float* w_ih  = (const float*)d_in[1];   // [512, 256]
    const float* w_hh  = (const float*)d_in[2];   // [512, 512]
    const float* b_h   = (const float*)d_in[3];   // [512]
    const float* w_out = (const float*)d_in[4];   // [256, 512]
    const float* b_out = (const float*)d_in[5];   // [256]
    float*       out   = (float*)d_out;           // [64, 2048, 256]

    char* ws = (char*)d_ws;
    __bf16* w_ih_b  = (__bf16*)(ws);                                    // 256 KB
    __bf16* w_hh_b  = (__bf16*)(ws + (256 << 10));                      // 512 KB
    __bf16* w_out_b = (__bf16*)(ws + (768 << 10));                      // 256 KB
    float*  x_proj  = (float*) (ws + (1 << 20));                        // 256 MB
    __bf16* hs      = (__bf16*)(ws + (1 << 20) + (size_t)BT * HID * 4); // 128 MB

    cvt_weights<<<(HID * HID + 255) / 256, 256, 0, stream>>>(w_ih, w_hh, w_out,
                                                             w_ih_b, w_hh_b, w_out_b);
    xproj_kernel<<<8192, 256, 0, stream>>>(input, w_ih_b, b_h, x_proj);
    rnn_kernel<<<1, 1024, HSH_BYTES + XPS_BYTES, stream>>>(x_proj, w_hh_b, hs);
    out_kernel<<<4096, 256, 0, stream>>>(hs, w_out_b, b_out, out);
}
